// Model1_GAT_63264868270718
// MI455X (gfx1250) — matrix operand order repaired and checked by reading, not compiled, not run on hardware
//
#include <hip/hip_runtime.h>
#include <hip/hip_bf16.h>
#include <math.h>

#define NN 20000
#define EE 320000
#define GG 64
#define ETOT (EE + NN)

typedef __attribute__((ext_vector_type(16))) __bf16 v16bf;
typedef __attribute__((ext_vector_type(8)))  __bf16 v8bf;
typedef __attribute__((ext_vector_type(8)))  float  v8f;
typedef __attribute__((ext_vector_type(4)))  unsigned u32x4;
typedef __attribute__((ext_vector_type(8)))  unsigned u32x8;

__device__ __forceinline__ float gelu_exact(float x) {
    return 0.5f * x * (1.0f + erff(x * 0.70710678118654752f));
}
// order-preserving float<->uint encoding for atomicMax on floats
__device__ __forceinline__ unsigned fenc(float f) {
    unsigned u = __float_as_uint(f);
    return (u & 0x80000000u) ? ~u : (u | 0x80000000u);
}
__device__ __forceinline__ float fdec(unsigned u) {
    u = (u & 0x80000000u) ? (u & 0x7FFFFFFFu) : ~u;
    return __uint_as_float(u);
}

__global__ void fill_f32(float* p, float v, long n) {
    long i = (long)blockIdx.x * blockDim.x + threadIdx.x;
    if (i < n) p[i] = v;
}
__global__ void fill_u32(unsigned* p, unsigned v, long n) {
    long i = (long)blockIdx.x * blockDim.x + threadIdx.x;
    if (i < n) p[i] = v;
}

// ---- bf16 staging: pad K to Kp, zero-fill ----
__global__ void cvt_bf16_pad(const float* __restrict__ A, __bf16* __restrict__ out,
                             int M, int K, int Kp) {
    long idx = (long)blockIdx.x * blockDim.x + threadIdx.x;
    if (idx >= (long)M * Kp) return;
    int m = (int)(idx / Kp), k = (int)(idx % Kp);
    out[idx] = (__bf16)((k < K) ? A[(long)m * K + k] : 0.0f);
}
// ---- bf16 staging: transpose W[K,Nc] -> Wt[Nc,Kp], zero-pad K ----
__global__ void cvt_bf16_transpose(const float* __restrict__ W, __bf16* __restrict__ Wt,
                                   int K, int Nc, int Kp) {
    long idx = (long)blockIdx.x * blockDim.x + threadIdx.x;
    if (idx >= (long)Nc * Kp) return;
    int n = (int)(idx / Kp), k = (int)(idx % Kp);
    Wt[idx] = (__bf16)((k < K) ? W[(long)k * Nc + n] : 0.0f);
}

// ---- self-loop attr = mean of incoming edge attrs ----
__global__ void loop_attr_accum(const float* __restrict__ ea, const int* __restrict__ dstp,
                                float* __restrict__ sum, float* __restrict__ cnt) {
    int e = blockIdx.x * blockDim.x + threadIdx.x;
    if (e >= EE) return;
    int d = dstp[e];
    atomicAdd(&cnt[d], 1.0f);
#pragma unroll
    for (int i = 0; i < 7; ++i) atomicAdd(&sum[(long)d * 7 + i], ea[(long)e * 7 + i]);
}
__global__ void loop_attr_div(float* __restrict__ sum, const float* __restrict__ cnt) {
    int n = blockIdx.x * blockDim.x + threadIdx.x;
    if (n >= NN) return;
    float c = fmaxf(cnt[n], 1.0f);
#pragma unroll
    for (int i = 0; i < 7; ++i) sum[(long)n * 7 + i] /= c;
}

// Issue a TDM load of a 64-row x KC-col bf16 strip of Wt (row stride Kp) into LDS.
// D# built per CDNA5 ISA (2-group form, 2-D tile, data_size=2B).
__device__ __forceinline__ void tdm_load_strip(const __bf16* gsrc, unsigned lds_bytes,
                                               int KC, int Kp) {
    unsigned long long ga = (unsigned long long)(size_t)gsrc;
    u32x4 g0; u32x8 g1;
    g0[0] = 1u;                                   // count=1, user mode
    g0[1] = lds_bytes;                            // lds_addr (bytes)
    g0[2] = (unsigned)ga;                         // global_addr[31:0]
    g0[3] = (unsigned)((ga >> 32) & 0x1FFFFFFull) // global_addr[56:32]
            | (2u << 30);                         // type=2 ("image")
    g1[0] = 1u << 16;                             // data_size=1 (2 bytes)
    g1[1] = ((unsigned)KC & 0xFFFFu) << 16;       // tensor_dim0[15:0] = KC
    g1[2] = 64u << 16;                            // tensor_dim1 = 64 rows
    g1[3] = ((unsigned)KC) << 16;                 // tile_dim0 = KC
    g1[4] = 64u;                                  // tile_dim1 = 64
    g1[5] = (unsigned)Kp;                         // tensor_dim0_stride = Kp
    g1[6] = 0u;
    g1[7] = 0u;
    asm volatile("tensor_load_to_lds %0, %1" :: "s"(g0), "s"(g1) : "memory");
}

// ---- WMMA bf16 GEMM with double-buffered TDM weight staging ----
// Block = 8 m-tiles (16 rows each) x one 64-wide n-strip. Wave 0 DMAs the
// 64 x KC bf16 weight strip into LDS with tensor_load_to_lds; while all 8
// waves compute on buffer `cur`, the next K-chunk streams into the other
// buffer (TENSORcnt drained by wave 0 right before the flip barrier).
// A fragments stream from global (two b128 per lane); B fragments come from
// LDS (two ds_load_b128 per tile), all preloaded before the 4-WMMA burst.
// Clamped waves never exit early -> barriers + EXEC all-ones at WMMA.
__global__ __launch_bounds__(256) void gemm_wmma_bf16(
    const __bf16* __restrict__ Abf, const __bf16* __restrict__ Wt,
    const float* __restrict__ bias, float* __restrict__ Cout,
    int M, int Kp, int Nc) {
    extern __shared__ __bf16 Bls[];              // 2 buffers of 64 * KC bf16
    int lane = threadIdx.x & 31;
    int wave = threadIdx.x >> 5;
    int tiles_m = (M + 15) >> 4;
    int gm = (tiles_m + 7) >> 3;                 // 8 m-tiles per block
    int bm = blockIdx.x % gm, bn = blockIdx.x / gm;
    int tmi = bm * 8 + wave;
    bool valid = tmi < tiles_m;
    int tm = valid ? tmi : tiles_m - 1;          // clamp; guard stores later
    int row0 = tm << 4, col0 = bn << 6;
    int half = lane >> 4, r = lane & 15;
    const __bf16* Ap = Abf + (long)(row0 + r) * Kp + half * 8;
    unsigned lds0 = (unsigned)(size_t)&Bls[0];   // flat low 32 bits = LDS byte offset
    const int KC = (Kp < 256) ? Kp : 256;
    const unsigned bufBytes = (unsigned)(64 * KC * 2);
    const __bf16* Wcol = Wt + (long)col0 * Kp;
    int nch = Kp / KC;

    // prologue: DMA chunk 0 into buffer 0
    if (wave == 0) {
        tdm_load_strip(Wcol, lds0, KC, Kp);
        __builtin_amdgcn_s_wait_tensorcnt(0);
    }
    __syncthreads();

    v8f acc[4] = {{}, {}, {}, {}};
    int cur = 0;
    for (int ci = 0; ci < nch; ++ci) {
        int kc = ci * KC;
        if (wave == 0 && ci + 1 < nch)           // overlap: prefetch next chunk
            tdm_load_strip(Wcol + kc + KC, lds0 + (cur ^ 1) * bufBytes, KC, Kp);
        const __bf16* Bbase = Bls + (long)cur * 64 * KC;
#pragma unroll 2
        for (int k0 = 0; k0 < KC; k0 += 32) {
            __builtin_prefetch(Ap + kc + k0 + 32, 0, 1);
            v8bf alo = *(const v8bf*)(Ap + kc + k0);
            v8bf ahi = *(const v8bf*)(Ap + kc + k0 + 16);
            v16bf af;
#pragma unroll
            for (int e = 0; e < 8; ++e) { af[e] = alo[e]; af[8 + e] = ahi[e]; }
            v8bf bs0[4], bs1[4];                 // preload all B fragments
#pragma unroll
            for (int t = 0; t < 4; ++t) {
                const __bf16* bp = Bbase + (long)(t * 16 + r) * KC + half * 8 + k0;
                bs0[t] = *(const v8bf*)bp;
                bs1[t] = *(const v8bf*)(bp + 16);
            }
#pragma unroll
            for (int t = 0; t < 4; ++t) {
                v16bf bfr;
#pragma unroll
                for (int e = 0; e < 8; ++e) { bfr[e] = bs0[t][e]; bfr[8 + e] = bs1[t][e]; }
                acc[t] = __builtin_amdgcn_wmma_f32_16x16x32_bf16(false, af, false, bfr,
                                                                 (short)0, acc[t], false, false);
            }
        }
        if (wave == 0) __builtin_amdgcn_s_wait_tensorcnt(0);   // next buffer landed
        __syncthreads();                                       // flip
        cur ^= 1;
    }
    if (valid) {
#pragma unroll
        for (int t = 0; t < 4; ++t) {
#pragma unroll
            for (int i = 0; i < 8; ++i) {
                int m = row0 + half * 8 + i;   // C/D: VGPR i -> M=i (lo) / M=i+8 (hi)
                int n = col0 + t * 16 + r;
                if (m < M) Cout[(long)m * Nc + n] = acc[t][i] + bias[n];
            }
        }
    }
}

// ---- attention logits + segment max (em = ea@We computed on the fly) ----
__global__ void gat_logits(const float* __restrict__ xl, const float* __restrict__ xr,
                           const float* __restrict__ ea, const float* __restrict__ lattr,
                           const int* __restrict__ src, const int* __restrict__ dstp,
                           const float* __restrict__ We, const float* __restrict__ att,
                           float* __restrict__ alog, unsigned* __restrict__ amax,
                           int H, int C) {
    long idx = (long)blockIdx.x * blockDim.x + threadIdx.x;
    if (idx >= (long)ETOT * H) return;
    int e = (int)(idx / H), hd = (int)(idx % H);
    int s, d; const float* ev;
    if (e < EE) { s = src[e]; d = dstp[e]; ev = ea + (long)e * 7; }
    else        { s = d = e - EE;          ev = lattr + (long)s * 7; }
    float a0 = ev[0], a1 = ev[1], a2 = ev[2], a3 = ev[3], a4 = ev[4], a5 = ev[5], a6 = ev[6];
    int HC = H * C;
    const float* xs = xl + (long)s * HC + hd * C;
    const float* xd = xr + (long)d * HC + hd * C;
    const float* Wc = We + hd * C;        // row stride HC
    const float* at = att + hd * C;
    float acc = 0.f;
    for (int c = 0; c < C; ++c) {
        float em = a0 * Wc[c] + a1 * Wc[HC + c] + a2 * Wc[2 * HC + c] + a3 * Wc[3 * HC + c]
                 + a4 * Wc[4 * HC + c] + a5 * Wc[5 * HC + c] + a6 * Wc[6 * HC + c];
        float m = xs[c] + xd[c] + em;
        m = (m > 0.f) ? m : 0.2f * m;     // LeakyReLU(0.2)
        acc += m * at[c];
    }
    alog[idx] = acc;
    atomicMax(&amax[(long)d * H + hd], fenc(acc));
}

__global__ void gat_exp(float* __restrict__ alog, const unsigned* __restrict__ amax,
                        float* __restrict__ den, const int* __restrict__ dstp, int H) {
    long idx = (long)blockIdx.x * blockDim.x + threadIdx.x;
    if (idx >= (long)ETOT * H) return;
    int e = (int)(idx / H), hd = (int)(idx % H);
    int d = (e < EE) ? dstp[e] : e - EE;
    float ex = expf(alog[idx] - fdec(amax[(long)d * H + hd]));
    alog[idx] = ex;
    atomicAdd(&den[(long)d * H + hd], ex);
}

__global__ void gat_aggregate(const float* __restrict__ ex, const float* __restrict__ den,
                              const float* __restrict__ xl, const int* __restrict__ src,
                              const int* __restrict__ dstp, float* __restrict__ agg,
                              int H, int C) {
    long idx = (long)blockIdx.x * blockDim.x + threadIdx.x;
    if (idx >= (long)ETOT * H) return;
    int e = (int)(idx / H), hd = (int)(idx % H);
    int s, d;
    if (e < EE) { s = src[e]; d = dstp[e]; } else { s = d = e - EE; }
    float alpha = ex[idx] / (den[(long)d * H + hd] + 1e-16f);
    const float* xs = xl + (long)s * H * C + hd * C;
    float*       ag = agg + (long)d * H * C + hd * C;
    for (int c = 0; c < C; ++c) atomicAdd(&ag[c], alpha * xs[c]);
}

__global__ void finalize_gelu(const float* __restrict__ agg, const float* __restrict__ bias,
                              float* __restrict__ nxt, int H, int C, int concat) {
    int OD = concat ? H * C : C;
    long idx = (long)blockIdx.x * blockDim.x + threadIdx.x;
    if (idx >= (long)NN * OD) return;
    int n = (int)(idx / OD), j = (int)(idx % OD);
    float y;
    if (concat) y = agg[(long)n * H * C + j] + bias[j];
    else {
        float s = 0.f;
        for (int h = 0; h < H; ++h) s += agg[(long)n * H * C + h * C + j];
        y = s / (float)H + bias[j];
    }
    nxt[idx] = gelu_exact(y);
}

// ---- graph readout: mean / max / sum over batch segments ----
__global__ void readout_accum(const float* __restrict__ h, const int* __restrict__ batch,
                              float* __restrict__ nper, float* __restrict__ addv,
                              unsigned* __restrict__ mxe) {
    long i = (long)blockIdx.x * blockDim.x + threadIdx.x;
    if (i >= (long)NN * 64) return;
    int n = (int)(i >> 6), d = (int)(i & 63);
    int g = batch[n];
    float v = h[i];
    atomicAdd(&addv[g * 64 + d], v);
    atomicMax(&mxe[g * 64 + d], fenc(v));
    if (d == 0) atomicAdd(&nper[g], 1.0f);
}
__global__ void build_z(const float* __restrict__ nper, const float* __restrict__ addv,
                        const unsigned* __restrict__ mxe, float* __restrict__ z) {
    int i = blockIdx.x * blockDim.x + threadIdx.x;
    if (i >= GG * 64) return;
    int g = i >> 6, d = i & 63;
    float np = fmaxf(nper[g], 1.0f);
    float ad = addv[i];
    z[g * 192 + d]       = ad / np;
    z[g * 192 + 64 + d]  = fdec(mxe[i]);
    z[g * 192 + 128 + d] = ad;
}

// ---- MLP head: linear (+ GELU + LayerNorm); one block per row (G=64 rows) ----
__global__ void head_layer(const float* __restrict__ in, const float* __restrict__ W,
                           const float* __restrict__ b, const float* __restrict__ gam,
                           const float* __restrict__ bet, float* __restrict__ out,
                           int K, int Nout, int doGLN) {
    __shared__ float sv[384];
    __shared__ float smu, srstd;
    int row = blockIdx.x;
    for (int j = threadIdx.x; j < Nout; j += blockDim.x) {
        float acc = b[j];
        for (int k = 0; k < K; ++k) acc += in[row * K + k] * W[k * Nout + j];
        if (doGLN) acc = gelu_exact(acc);
        sv[j] = acc;
    }
    __syncthreads();
    if (doGLN) {
        if (threadIdx.x == 0) {
            float s = 0.f;
            for (int j = 0; j < Nout; ++j) s += sv[j];
            float mu = s / (float)Nout, v = 0.f;
            for (int j = 0; j < Nout; ++j) { float dv = sv[j] - mu; v += dv * dv; }
            smu = mu; srstd = rsqrtf(v / (float)Nout + 1e-5f);
        }
        __syncthreads();
        for (int j = threadIdx.x; j < Nout; j += blockDim.x)
            out[row * Nout + j] = (sv[j] - smu) * srstd * gam[j] + bet[j];
    } else {
        for (int j = threadIdx.x; j < Nout; j += blockDim.x) out[row * Nout + j] = sv[j];
    }
}

extern "C" void kernel_launch(void* const* d_in, const int* in_sizes, int n_in,
                              void* d_out, int out_size, void* d_ws, size_t ws_size,
                              hipStream_t stream) {
    (void)in_sizes; (void)n_in; (void)out_size; (void)ws_size;
    const float* x     = (const float*)d_in[0];
    const float* ea    = (const float*)d_in[1];
    const int*   ei    = (const int*)d_in[2];
    const int*   batch = (const int*)d_in[3];
    const int* src  = ei;
    const int* dstp = ei + EE;

    // workspace carve-up (float units; all chunks 16B-aligned)
    float* ws = (float*)d_ws;
    float* bufA  = ws; ws += (long)NN * 768;
    float* bufB  = ws; ws += (long)NN * 768;
    float* xl    = ws; ws += (long)NN * 768;
    float* xr    = ws; ws += (long)NN * 768;   // doubles as aggregation target
    float* lattr = ws; ws += (long)NN * 8;     // 7 used, 8 for alignment
    float* cnt   = ws; ws += NN;
    float* alog  = ws; ws += (long)ETOT * 6;
    float* den   = ws; ws += (long)NN * 6;
    unsigned* amax = (unsigned*)ws; ws += (long)NN * 6;
    float* nper  = ws; ws += GG;
    float* addv  = ws; ws += (long)GG * 64;
    unsigned* mxe = (unsigned*)ws; ws += (long)GG * 64;
    float* z0    = ws; ws += (long)GG * 384;
    float* z1    = ws; ws += (long)GG * 384;
    ws += ((16 - ((z1 - (float*)d_ws) & 15)) & 15);          // 64B align bf16 region
    __bf16* Abf = (__bf16*)ws; ws += (long)NN * 768 / 2;     // [M, Kp] bf16
    __bf16* Wtl = (__bf16*)ws; ws += (long)768 * 768 / 2;    // [Nc, Kp] bf16
    __bf16* Wtr = (__bf16*)ws; ws += (long)768 * 768 / 2;

    auto cdiv = [](long a, long b) { return (int)((a + b - 1) / b); };
    const unsigned NEGMAX = 0x00800000u;   // fenc(-FLT_MAX)

    // self-loop edge attrs (fill_value='mean')
    fill_f32<<<cdiv((long)NN * 7, 256), 256, 0, stream>>>(lattr, 0.f, (long)NN * 7);
    fill_f32<<<cdiv(NN, 256), 256, 0, stream>>>(cnt, 0.f, (long)NN);
    loop_attr_accum<<<cdiv(EE, 256), 256, 0, stream>>>(ea, dstp, lattr, cnt);
    loop_attr_div<<<cdiv(NN, 256), 256, 0, stream>>>(lattr, cnt);

    struct LCfg { int K, C, H, concat; };
    LCfg L[4] = {{14, 128, 6, 1}, {768, 128, 6, 1}, {768, 64, 4, 1}, {256, 64, 2, 0}};
    const float* cur = x;
    float* nxtbuf[4] = {bufA, bufB, bufA, bufB};

    for (int li = 0; li < 4; ++li) {
        int base = 4 + li * 7;
        const float* Wl   = (const float*)d_in[base + 0];
        const float* bl   = (const float*)d_in[base + 1];
        const float* Wr   = (const float*)d_in[base + 2];
        const float* br   = (const float*)d_in[base + 3];
        const float* We   = (const float*)d_in[base + 4];
        const float* att  = (const float*)d_in[base + 5];
        const float* bias = (const float*)d_in[base + 6];
        int K = L[li].K, C = L[li].C, H = L[li].H, HC = H * C;
        int Kp = ((K + 31) / 32) * 32;
        int KC = (Kp < 256) ? Kp : 256;
        size_t ldsB = (size_t)2 * 64 * KC * 2;   // double-buffered strip

        // stage to bf16 (padded A, transposed+padded weights)
        cvt_bf16_pad<<<cdiv((long)NN * Kp, 256), 256, 0, stream>>>(cur, Abf, NN, K, Kp);
        cvt_bf16_transpose<<<cdiv((long)HC * Kp, 256), 256, 0, stream>>>(Wl, Wtl, K, HC, Kp);
        cvt_bf16_transpose<<<cdiv((long)HC * Kp, 256), 256, 0, stream>>>(Wr, Wtr, K, HC, Kp);

        int tiles_m = (NN + 15) / 16;
        int gm = (tiles_m + 7) / 8;
        int gb = gm * (HC / 64);
        gemm_wmma_bf16<<<gb, 256, ldsB, stream>>>(Abf, Wtl, bl, xl, NN, Kp, HC);
        gemm_wmma_bf16<<<gb, 256, ldsB, stream>>>(Abf, Wtr, br, xr, NN, Kp, HC);

        fill_u32<<<cdiv((long)NN * H, 256), 256, 0, stream>>>(amax, NEGMAX, (long)NN * H);
        fill_f32<<<cdiv((long)NN * H, 256), 256, 0, stream>>>(den, 0.f, (long)NN * H);
        gat_logits<<<cdiv((long)ETOT * H, 256), 256, 0, stream>>>(
            xl, xr, ea, lattr, src, dstp, We, att, alog, amax, H, C);
        gat_exp<<<cdiv((long)ETOT * H, 256), 256, 0, stream>>>(alog, amax, den, dstp, H);

        // xr is free after logits: reuse as aggregation target
        fill_f32<<<cdiv((long)NN * HC, 256), 256, 0, stream>>>(xr, 0.f, (long)NN * HC);
        gat_aggregate<<<cdiv((long)ETOT * H, 256), 256, 0, stream>>>(
            alog, den, xl, src, dstp, xr, H, C);

        int OD = L[li].concat ? HC : C;
        finalize_gelu<<<cdiv((long)NN * OD, 256), 256, 0, stream>>>(
            xr, bias, nxtbuf[li], H, C, L[li].concat);
        cur = nxtbuf[li];
    }

    // readout over batch segments
    fill_f32<<<1, GG, 0, stream>>>(nper, 0.f, GG);
    fill_f32<<<cdiv((long)GG * 64, 256), 256, 0, stream>>>(addv, 0.f, (long)GG * 64);
    fill_u32<<<cdiv((long)GG * 64, 256), 256, 0, stream>>>(mxe, NEGMAX, (long)GG * 64);
    readout_accum<<<cdiv((long)NN * 64, 256), 256, 0, stream>>>(cur, batch, nper, addv, mxe);
    build_z<<<cdiv((long)GG * 64, 256), 256, 0, stream>>>(nper, addv, mxe, z0);

    // head MLP
    int hk[5] = {192, 384, 256, 128, 64};
    int hn[5] = {384, 256, 128, 64, 9};
    float* hin = z0; float* hout = z1;
    for (int hi = 0; hi < 5; ++hi) {
        int pb = 32 + hi * 4;
        const float* W  = (const float*)d_in[pb];
        const float* b  = (const float*)d_in[pb + 1];
        const float* g  = (hi < 4) ? (const float*)d_in[pb + 2] : nullptr;
        const float* be = (hi < 4) ? (const float*)d_in[pb + 3] : nullptr;
        float* outp = (hi == 4) ? (float*)d_out : hout;
        head_layer<<<GG, 128, 0, stream>>>(hin, W, b, g, be, outp, hk[hi], hn[hi], hi < 4 ? 1 : 0);
        float* t = hin; hin = hout; hout = t;
    }
}
